// MultiHeadAttention_31233002176665
// MI455X (gfx1250) — matrix operand order repaired and checked by reading, not compiled, not run on hardware
//
#include <hip/hip_runtime.h>
#include <hip/hip_bf16.h>

// ---------------------------------------------------------------------------
// Problem constants (from reference)
// ---------------------------------------------------------------------------
#define NA_N 50000
#define NB_N 50000
#define E_N  800000
#define CA_N 256
#define CB_N 256
#define D_N  256
#define H_N  8
#define DH_N 32

typedef __attribute__((ext_vector_type(16))) __bf16 v16bf;
typedef __attribute__((ext_vector_type(8)))  float  v8f;

__device__ __forceinline__ __bf16 f2bf(float f) { return (__bf16)f; }

// ---------------------------------------------------------------------------
// WMMA fragment loaders for v_wmma_f32_16x16x32_bf16 (contiguous k order, shared by A and B)
// ---------------------------------------------------------------------------
// A: 16x32 (MxK) bf16. Lane l<16: M=l, slot s -> K=s.
//    Lane l>=16: M=l-16, slot s -> K=16+s.
// Source matrix is f32 row-major (lda), converted on the fly.
__device__ __forceinline__ v16bf load_a_frag_f32(const float* A, int lda,
                                                 int m0, int k0, int lane) {
    int laneM = lane & 15;
    int hi    = lane >> 4;
    const float* arow = A + (size_t)(m0 + laneM) * lda + k0;
    const float4* p0 = (const float4*)(arow + hi * 16);       // K = hi*16 .. +7
    const float4* p1 = (const float4*)(arow + hi * 16 + 8);   // K = hi*16+8 .. +7
    float4 x0 = p0[0], x1 = p0[1];
    float4 y0 = p1[0], y1 = p1[1];
    v16bf a;
    a[0]  = f2bf(x0.x); a[1]  = f2bf(x0.y); a[2]  = f2bf(x0.z); a[3]  = f2bf(x0.w);
    a[4]  = f2bf(x1.x); a[5]  = f2bf(x1.y); a[6]  = f2bf(x1.z); a[7]  = f2bf(x1.w);
    a[8]  = f2bf(y0.x); a[9]  = f2bf(y0.y); a[10] = f2bf(y0.z); a[11] = f2bf(y0.w);
    a[12] = f2bf(y1.x); a[13] = f2bf(y1.y); a[14] = f2bf(y1.z); a[15] = f2bf(y1.w);
    return a;
}

// B: 32x16 (KxN) bf16, B[k][n] = W[n0+n][k0+k] for y = x @ W^T.
// Lane l<16: N=l, slot s -> K=s.  Lane l>=16: N=l-16, slot s -> K=16+s.
// => 16 contiguous bf16 (32B, aligned) per lane from the weight row.
__device__ __forceinline__ v16bf load_b_frag_bf16(const __bf16* W, int ldw,
                                                  int n0, int k0, int lane) {
    int laneN = lane & 15;
    int kb    = (lane >> 4) * 16;
    const __bf16* wrow = W + (size_t)(n0 + laneN) * ldw + k0 + kb;
    return *(const v16bf*)wrow;
}

#define WMMA_BF16(A, B, C) \
    __builtin_amdgcn_wmma_f32_16x16x32_bf16(false, (A), false, (B), (short)0, (C), false, false)

// ---------------------------------------------------------------------------
// Kernel 1: convert the four 256x256 weight matrices f32 -> bf16
// ---------------------------------------------------------------------------
__global__ void k_cvt_w(const float* __restrict__ w0, const float* __restrict__ w1,
                        const float* __restrict__ w2, const float* __restrict__ w3,
                        __bf16* o0, __bf16* o1, __bf16* o2, __bf16* o3) {
    int i = blockIdx.x * 256 + threadIdx.x;   // grid covers 65536 elems
    o0[i] = f2bf(w0[i]);
    o1[i] = f2bf(w1[i]);
    o2[i] = f2bf(w2[i]);
    o3[i] = f2bf(w3[i]);
}

// ---------------------------------------------------------------------------
// Kernel 2: init attn accumulator = 0, segmax = -inf (bits), segsum = 0
// ---------------------------------------------------------------------------
__global__ void k_init(float* attn, unsigned int* segmax_bits, float* segsum) {
    size_t i = (size_t)blockIdx.x * 256 + threadIdx.x;   // grid covers NA*256
    attn[i] = 0.0f;
    if (i < (size_t)NA_N * H_N) {
        segmax_bits[i] = 0xFF800000u;   // -inf
        segsum[i]      = 0.0f;
    }
}

// ---------------------------------------------------------------------------
// Kernel 3: Q = Fa @ Wq^T   (one 16-row M-tile per block, 8 waves x 32 cols)
// ---------------------------------------------------------------------------
__global__ void k_gemm_x1(const float* __restrict__ A, const __bf16* __restrict__ W,
                          float* __restrict__ C) {
    int lane = threadIdx.x & 31;
    int wid  = threadIdx.x >> 5;         // 0..7
    int m0   = blockIdx.x * 16;
    int n0   = wid * 32;
    v8f acc0 = {}; v8f acc1 = {};
    for (int k0 = 0; k0 < CA_N; k0 += 32) {
        v16bf af = load_a_frag_f32(A, CA_N, m0, k0, lane);
        v16bf b0 = load_b_frag_bf16(W, CA_N, n0, k0, lane);
        v16bf b1 = load_b_frag_bf16(W, CA_N, n0 + 16, k0, lane);
        acc0 = WMMA_BF16(af, b0, acc0);
        acc1 = WMMA_BF16(af, b1, acc1);
    }
    int laneN = lane & 15;
    int mo    = (lane >> 4) * 8;
    for (int r = 0; r < 8; ++r) {
        size_t row = (size_t)(m0 + mo + r) * D_N;
        C[row + n0 + laneN]      = acc0[r];
        C[row + n0 + 16 + laneN] = acc1[r];
    }
}

// ---------------------------------------------------------------------------
// Kernel 4: fused K/V GEMM — A fragment reused across both weight matrices
// ---------------------------------------------------------------------------
__global__ void k_gemm_x2(const float* __restrict__ A,
                          const __bf16* __restrict__ Wk, const __bf16* __restrict__ Wv,
                          float* __restrict__ Ko, float* __restrict__ Vo) {
    int lane = threadIdx.x & 31;
    int wid  = threadIdx.x >> 5;
    int m0   = blockIdx.x * 16;
    int n0   = wid * 32;
    v8f ak0 = {}, ak1 = {}, av0 = {}, av1 = {};
    for (int k0 = 0; k0 < CB_N; k0 += 32) {
        v16bf af  = load_a_frag_f32(A, CB_N, m0, k0, lane);
        v16bf bk0 = load_b_frag_bf16(Wk, CB_N, n0, k0, lane);
        v16bf bk1 = load_b_frag_bf16(Wk, CB_N, n0 + 16, k0, lane);
        v16bf bv0 = load_b_frag_bf16(Wv, CB_N, n0, k0, lane);
        v16bf bv1 = load_b_frag_bf16(Wv, CB_N, n0 + 16, k0, lane);
        ak0 = WMMA_BF16(af, bk0, ak0);
        ak1 = WMMA_BF16(af, bk1, ak1);
        av0 = WMMA_BF16(af, bv0, av0);
        av1 = WMMA_BF16(af, bv1, av1);
    }
    int laneN = lane & 15;
    int mo    = (lane >> 4) * 8;
    for (int r = 0; r < 8; ++r) {
        size_t row = (size_t)(m0 + mo + r) * D_N;
        Ko[row + n0 + laneN]      = ak0[r];
        Ko[row + n0 + 16 + laneN] = ak1[r];
        Vo[row + n0 + laneN]      = av0[r];
        Vo[row + n0 + 16 + laneN] = av1[r];
    }
}

// ---------------------------------------------------------------------------
// Kernel 5: per-edge logits (lane = dh, shuffle reduce) + float atomic max
// ---------------------------------------------------------------------------
__global__ void k_edge_logits(const float* __restrict__ Q, const float* __restrict__ K,
                              const int* __restrict__ a_idx, const int* __restrict__ b_idx,
                              float* __restrict__ logits, unsigned int* __restrict__ segmax) {
    int lane = threadIdx.x & 31;
    int wid  = threadIdx.x >> 5;
    size_t e = (size_t)blockIdx.x * 8 + wid;          // E divisible by 8
    int a = a_idx[e], b = b_idx[e];
    const float* qrow = Q + (size_t)a * D_N;
    const float* krow = K + (size_t)b * D_N;
    const float scale = 0.17677669529663687f;         // 1/sqrt(DH)
    for (int h = 0; h < H_N; ++h) {
        float p = qrow[h * DH_N + lane] * krow[h * DH_N + lane];
        p += __shfl_xor(p, 16);
        p += __shfl_xor(p, 8);
        p += __shfl_xor(p, 4);
        p += __shfl_xor(p, 2);
        p += __shfl_xor(p, 1);
        if (lane == 0) {
            float lg = p * scale;
            logits[e * H_N + h] = lg;
            unsigned int* addr = segmax + (size_t)a * H_N + h;
            // float-ordered atomic max via signed/unsigned integer atomics
            if (lg >= 0.0f) atomicMax((int*)addr, __float_as_int(lg));
            else            atomicMin(addr, __float_as_uint(lg));
        }
    }
}

// ---------------------------------------------------------------------------
// Kernel 6: ex = exp(logit - segmax); segsum += ex (in-place on logits buffer)
// ---------------------------------------------------------------------------
__global__ void k_edge_exp(const int* __restrict__ a_idx,
                           const float* __restrict__ segmax,
                           float* __restrict__ exbuf, float* __restrict__ segsum) {
    size_t i = (size_t)blockIdx.x * 256 + threadIdx.x;  // covers E*H exactly
    size_t e = i >> 3;
    int    h = (int)(i & 7);
    int    a = a_idx[e];
    float  m = segmax[(size_t)a * H_N + h];             // finite: includes this edge
    float ex = __expf(exbuf[i] - m);
    exbuf[i] = ex;
    atomicAdd(segsum + (size_t)a * H_N + h, ex);
}

// ---------------------------------------------------------------------------
// Kernel 7: scatter-add weighted V rows into per-query accumulator
// ---------------------------------------------------------------------------
__global__ void k_edge_scatter(const float* __restrict__ V,
                               const int* __restrict__ a_idx, const int* __restrict__ b_idx,
                               const float* __restrict__ exbuf, const float* __restrict__ segsum,
                               float* __restrict__ attn) {
    int lane = threadIdx.x & 31;
    int wid  = threadIdx.x >> 5;
    size_t e = (size_t)blockIdx.x * 8 + wid;
    int a = a_idx[e], b = b_idx[e];
    const float* vrow = V + (size_t)b * D_N;
    float* orow = attn + (size_t)a * D_N;
    for (int h = 0; h < H_N; ++h) {
        float w = exbuf[e * H_N + h] / fmaxf(segsum[(size_t)a * H_N + h], 1e-30f);
        atomicAdd(orow + h * DH_N + lane, w * vrow[h * DH_N + lane]);
    }
}

// ---------------------------------------------------------------------------
// Kernel 8: out = Fa + attn @ Wproj^T   (WMMA, residual add in epilogue)
// ---------------------------------------------------------------------------
__global__ void k_gemm_proj(const float* __restrict__ attn, const __bf16* __restrict__ Wp,
                            const float* __restrict__ Fa, float* __restrict__ out) {
    int lane = threadIdx.x & 31;
    int wid  = threadIdx.x >> 5;
    int m0   = blockIdx.x * 16;
    int n0   = wid * 32;
    v8f acc0 = {}; v8f acc1 = {};
    for (int k0 = 0; k0 < D_N; k0 += 32) {
        v16bf af = load_a_frag_f32(attn, D_N, m0, k0, lane);
        v16bf b0 = load_b_frag_bf16(Wp, D_N, n0, k0, lane);
        v16bf b1 = load_b_frag_bf16(Wp, D_N, n0 + 16, k0, lane);
        acc0 = WMMA_BF16(af, b0, acc0);
        acc1 = WMMA_BF16(af, b1, acc1);
    }
    int laneN = lane & 15;
    int mo    = (lane >> 4) * 8;
    for (int r = 0; r < 8; ++r) {
        size_t row = (size_t)(m0 + mo + r) * CA_N;
        out[row + n0 + laneN]      = Fa[row + n0 + laneN]      + acc0[r];
        out[row + n0 + 16 + laneN] = Fa[row + n0 + 16 + laneN] + acc1[r];
    }
}

// ---------------------------------------------------------------------------
// Host launcher
// ---------------------------------------------------------------------------
extern "C" void kernel_launch(void* const* d_in, const int* in_sizes, int n_in,
                              void* d_out, int out_size, void* d_ws, size_t ws_size,
                              hipStream_t stream) {
    const float* Fa    = (const float*)d_in[0];
    const float* Fb    = (const float*)d_in[1];
    const int*   a_idx = (const int*)  d_in[2];
    const int*   b_idx = (const int*)  d_in[3];
    const float* Wq    = (const float*)d_in[4];
    const float* Wk    = (const float*)d_in[5];
    const float* Wv    = (const float*)d_in[6];
    const float* Wp    = (const float*)d_in[7];
    float* out = (float*)d_out;

    char* ws = (char*)d_ws;
    size_t off = 0;
    auto take = [&](size_t bytes) -> char* {
        char* p = ws + off;
        off += (bytes + 255) & ~(size_t)255;
        return p;
    };
    __bf16* Wq_bf = (__bf16*)take((size_t)D_N * CA_N * 2);
    __bf16* Wk_bf = (__bf16*)take((size_t)D_N * CB_N * 2);
    __bf16* Wv_bf = (__bf16*)take((size_t)D_N * CB_N * 2);
    __bf16* Wp_bf = (__bf16*)take((size_t)CA_N * D_N * 2);
    float*  Qb    = (float*)take((size_t)NA_N * D_N * 4);
    float*  Kb    = (float*)take((size_t)NB_N * D_N * 4);
    float*  Vb    = (float*)take((size_t)NB_N * D_N * 4);
    float*  attn  = (float*)take((size_t)NA_N * D_N * 4);
    float*  exb   = (float*)take((size_t)E_N * H_N * 4);
    unsigned int* segmax = (unsigned int*)take((size_t)NA_N * H_N * 4);
    float*  segsum = (float*)take((size_t)NA_N * H_N * 4);

    k_cvt_w<<<(D_N * CA_N) / 256, 256, 0, stream>>>(Wq, Wk, Wv, Wp,
                                                    Wq_bf, Wk_bf, Wv_bf, Wp_bf);
    k_init<<<(NA_N * D_N) / 256, 256, 0, stream>>>(attn, segmax, segsum);
    k_gemm_x1<<<NA_N / 16, 256, 0, stream>>>(Fa, Wq_bf, Qb);
    k_gemm_x2<<<NB_N / 16, 256, 0, stream>>>(Fb, Wk_bf, Wv_bf, Kb, Vb);
    k_edge_logits<<<E_N / 8, 256, 0, stream>>>(Qb, Kb, a_idx, b_idx, exb, segmax);
    k_edge_exp<<<(E_N * H_N) / 256, 256, 0, stream>>>(a_idx, (const float*)segmax, exb, segsum);
    k_edge_scatter<<<E_N / 8, 256, 0, stream>>>(Vb, a_idx, b_idx, exb, segsum, attn);
    k_gemm_proj<<<NA_N / 16, 256, 0, stream>>>(attn, Wp_bf, Fa, out);
}
